// VanillaRNN_63230508531997
// MI455X (gfx1250) — matrix operand order repaired and checked by reading, not compiled, not run on hardware
//
#include <hip/hip_runtime.h>

#define SEQ     256
#define HID     1024
#define CLASSES 10
#define NBT     16               // batch tiles (16 rows each)
#define NCG     8                // column groups (128 cols each)
#define RS      1032             // padded LDS row stride (bf16); row = 2064 B
#define HELEMS  (256 * HID)      // elements in one h buffer

typedef __attribute__((ext_vector_type(16))) __bf16 v16bf;
typedef __attribute__((ext_vector_type(8)))  __bf16 v8bf;
typedef __attribute__((ext_vector_type(8)))  float  v8f;
typedef __attribute__((ext_vector_type(4)))  unsigned vu4;

// branch-free tanh: 1 - 2/(e^{2x}+1); v_exp_f32/v_rcp_f32 saturate correctly
// at +/-inf so no range clamping needed. bf16 storage swallows the ~1ulp error.
__device__ __forceinline__ float fast_tanh(float v) {
    float e = __builtin_amdgcn_exp2f(v * 2.8853900817779268f);   // e^(2v)
    return 1.f - 2.f * __builtin_amdgcn_rcpf(e + 1.f);
}

// ---------------------------------------------------------------------------
// Pack W_hh (f32 [K=1024][N=1024]) into bf16 WMMA B-fragment order, grouped so
// each column-group's slice is one contiguous 256 KB block:
//   dst d = ((nt*32 + kc)*32 + l)*16 + i
//   K = kc*32 + (l>=16)*16 + i,   N = nt*16 + (l&15)      (contiguous k; A uses same)
// ---------------------------------------------------------------------------
__global__ void pack_whh_kernel(const float* __restrict__ W,
                                __bf16* __restrict__ Wp) {
    int d  = blockIdx.x * blockDim.x + threadIdx.x;
    int i  = d & 15;
    int l  = (d >> 4) & 31;
    int kc = (d >> 9) & 31;
    int nt = d >> 14;
    int K  = kc * 32 + ((l >> 4) << 4) + i;
    int N  = nt * 16 + (l & 15);
    Wp[d] = (__bf16)W[K * HID + N];
}

// Re-zero barrier counters and h buffer 0 every launch (ws is not re-poisoned).
__global__ void init_ws_kernel(unsigned* __restrict__ cnt,
                               unsigned* __restrict__ h0) {
    int i = blockIdx.x * blockDim.x + threadIdx.x;
    if (i < 512) cnt[i] = 0u;
    if (i < HELEMS / 2) h0[i] = 0u;          // 2 bf16 per uint
}

// ---------------------------------------------------------------------------
// Persistent RNN. Block (bt, cg): batch rows [16bt,16bt+16), cols [128cg,+128).
// 8 wave32s; wave w owns global N-tile cg*8+w. W slice + x tile live in LDS
// for the whole kernel; h is exchanged through L2 (async loads, SCOPE_DEV)
// with a per-batch-tile agent-scope atomic barrier.
// ---------------------------------------------------------------------------
__global__ __launch_bounds__(256, 1)
void rnn_persistent_kernel(const float* __restrict__ x,
                           const float* __restrict__ W_hx,
                           const float* __restrict__ W_ph,
                           const float* __restrict__ b_h,
                           const float* __restrict__ b_p,
                           const __bf16* __restrict__ Wp,
                           __bf16* __restrict__ hg,      // [2][256][1024] bf16
                           unsigned* __restrict__ cnt,   // [16] spaced by 32
                           float* __restrict__ out) {
    __shared__ __bf16 Wlds[8 * 32 * 512];   // 256 KB resident weights
    __shared__ __bf16 hlds[16 * RS];        // 33 KB h_{t-1} staging
    __shared__ float  xlds[16 * SEQ];       // 16 KB resident x tile

    const int tid  = threadIdx.x;
    const int lane = tid & 31;
    const int wave = tid >> 5;
    const int bt   = blockIdx.x >> 3;
    const int cg   = blockIdx.x & 7;

    // one-time: stage 256 KB weight slice + 16 KB x tile into LDS
    {
        const vu4* __restrict__ ws = (const vu4*)(Wp + (size_t)(cg * 8) * 32 * 512);
        vu4* wd = (vu4*)Wlds;
        for (int c = tid; c < 16384; c += 256) wd[c] = ws[c];
        const vu4* __restrict__ xs = (const vu4*)(x + (size_t)bt * 16 * SEQ);
        vu4* xd = (vu4*)xlds;
        for (int c = tid; c < 1024; c += 256) xd[c] = xs[c];
    }

    const int m    = lane & 15;             // A row this lane serves
    const int half = lane >> 4;
    const int off0 = half ? 16 : 0;
    const int col  = (cg * 8 + wave) * 16 + (lane & 15);
    const float whx = W_hx[col];
    const float bhv = b_h[col];

    unsigned* mycnt = cnt + bt * 32;        // one 128B line per batch tile
    int p = 0;                              // h double-buffer parity

    for (int t = 0; t < SEQ; ++t) {
        // stage h_{t-1}[bt] (16x1024 bf16 = 32 KB) into LDS via async
        // global->LDS DMA; SCOPE_DEV bypasses possibly-stale WGP$.
        {
            const char* gsrc =
                (const char*)(hg + (size_t)p * HELEMS + (size_t)bt * 16 * HID);
            for (int c = tid; c < 2048; c += 256) {
                int row = c >> 6, seg = c & 63;
                unsigned ldsa = (unsigned)(uintptr_t)&hlds[row * RS + seg * 8];
                const void* ga = gsrc + (size_t)c * 16;
                asm volatile(
                    "global_load_async_to_lds_b128 %0, %1, off scope:SCOPE_DEV"
                    :: "v"(ldsa), "v"(ga) : "memory");
            }
            asm volatile("s_wait_asynccnt 0x0" ::: "memory");
        }
        __syncthreads();   // also covers one-time Wlds/xlds fills at t==0

        v8f acc0 = (v8f){}, acc1 = (v8f){};
#pragma unroll 4
        for (int kc = 0; kc < 32; ++kc) {
            const int kb = kc * 32;
            v8bf lo = *(const v8bf*)&hlds[m * RS + kb + off0];
            v8bf hv = *(const v8bf*)&hlds[m * RS + kb + off0 + 8];
            v16bf a;
#pragma unroll
            for (int i = 0; i < 8; ++i) { a[i] = lo[i]; a[i + 8] = hv[i]; }
            const v16bf b =
                *(const v16bf*)&Wlds[((wave * 32 + kc) << 9) + (lane << 4)];
            if (kc & 1)
                acc1 = __builtin_amdgcn_wmma_f32_16x16x32_bf16(
                    false, a, false, b, (short)0, acc1, false, false);
            else
                acc0 = __builtin_amdgcn_wmma_f32_16x16x32_bf16(
                    false, a, false, b, (short)0, acc0, false, false);
        }

        // epilogue: h_t = tanh(acc + x[b,t]*W_hx + b_h) -> global (other buffer)
        __bf16* hdst = hg + (size_t)(p ^ 1) * HELEMS + (size_t)bt * 16 * HID;
#pragma unroll
        for (int r = 0; r < 8; ++r) {
            const int mr = r + half * 8;    // C-layout: VGPR r, lane half
            float v = acc0[r] + acc1[r] + xlds[mr * SEQ + t] * whx + bhv;
            hdst[mr * HID + col] = (__bf16)fast_tanh(v);
        }

        // sibling barrier: 8 column-group WGs of this batch tile
        __threadfence();                    // flush my stores to device scope
        __syncthreads();
        if (tid == 0) {
            __hip_atomic_fetch_add(mycnt, 1u, __ATOMIC_ACQ_REL,
                                   __HIP_MEMORY_SCOPE_AGENT);
            const unsigned target = (unsigned)(t + 1) * NCG;
            while (__hip_atomic_load(mycnt, __ATOMIC_ACQUIRE,
                                     __HIP_MEMORY_SCOPE_AGENT) < target)
                __builtin_amdgcn_s_sleep(2);
        }
        __syncthreads();
        p ^= 1;
    }

    // p toggled 256 times -> final h is buffer 0. Tiny 10-class projection.
    if (cg == 0 && tid < 16 * CLASSES) {
        const int mo = tid % 16;
        const int c  = tid / 16;
        const volatile unsigned short* hf =
            (const volatile unsigned short*)(hg + (size_t)bt * 16 * HID);
        float s = 0.f;
        for (int k = 0; k < HID; ++k) {
            unsigned bits = (unsigned)hf[mo * HID + k] << 16;
            s += __builtin_bit_cast(float, bits) * W_ph[k * CLASSES + c];
        }
        out[(bt * 16 + mo) * CLASSES + c] = s + b_p[c];
    }
}

extern "C" void kernel_launch(void* const* d_in, const int* in_sizes, int n_in,
                              void* d_out, int out_size, void* d_ws, size_t ws_size,
                              hipStream_t stream) {
    const float* x    = (const float*)d_in[0];  // [256, 256]
    const float* W_hx = (const float*)d_in[1];  // [1, 1024]
    const float* W_hh = (const float*)d_in[2];  // [1024, 1024]
    const float* W_ph = (const float*)d_in[3];  // [1024, 10]
    const float* b_h  = (const float*)d_in[4];  // [1024]
    const float* b_p  = (const float*)d_in[5];  // [10]
    float* out = (float*)d_out;                 // [256, 10]

    // workspace layout: [0,2KB) counters | [4KB, 4KB+1MB) h[2] | [2MB,4MB) Wp
    unsigned* cnt = (unsigned*)d_ws;
    __bf16*   hg  = (__bf16*)((char*)d_ws + 4096);
    __bf16*   Wp  = (__bf16*)((char*)d_ws + (2u << 20));

    init_ws_kernel<<<512, 256, 0, stream>>>(cnt, (unsigned*)hg);
    pack_whh_kernel<<<4096, 256, 0, stream>>>(W_hh, Wp);
    rnn_persistent_kernel<<<NBT * NCG, 256, 0, stream>>>(
        x, W_hx, W_ph, b_h, b_p, Wp, hg, cnt, out);
}
